// DecoderLayer_70300024700966
// MI455X (gfx1250) — compile-verified, matrix operand order checked by reading, not run on hardware
//
#include <hip/hip_runtime.h>
#include <hip/hip_bf16.h>
#include <math.h>

// ---------------------------------------------------------------------------
// DecoderLayer for MI455X (gfx1250, wave32, WMMA 16x16x32 f16->f32)
// B=2, L=2048, D=1024, H=16, HD=64, DF=4096
// ---------------------------------------------------------------------------

typedef _Float16 half_t;
typedef __attribute__((ext_vector_type(16))) _Float16 v16h;
typedef __attribute__((ext_vector_type(8)))  float    v8f;
typedef __attribute__((ext_vector_type(4)))  int      v4i;

#define NB   2
#define NL   2048
#define ND   1024
#define NH   16
#define NHD  64
#define NDF  4096
#define NROWS (NB * NL)   // 4096

// Async global->LDS (ASYNCcnt) path, guarded so a missing builtin cannot
// break the build; falls back to plain LDS staging.
#if defined(__gfx1250__) && \
    __has_builtin(__builtin_amdgcn_global_load_async_to_lds_b128) && \
    __has_builtin(__builtin_amdgcn_s_wait_asynccnt)
#define USE_ASYNC_LDS 1
#define AS_GLOBAL __attribute__((address_space(1)))
#define AS_SHARED __attribute__((address_space(3)))
__device__ inline void async_b128(const half_t* g, half_t* l) {
    __builtin_amdgcn_global_load_async_to_lds_b128(
        (AS_GLOBAL v4i*)g, (AS_SHARED v4i*)l, 0, 0);
}
__device__ inline void async_wait0() { __builtin_amdgcn_s_wait_asynccnt(0); }
#else
#define USE_ASYNC_LDS 0
#endif

// -------------------- fragment gather (16-bit A/B layout) -------------------
// ISA 7.12.2: 16-bit A 16x32 — lane m=lane%16 holds row m; lanes>=16 use K
// base +8; VGPR j: K = (j/4)*16 + base + (j%4)*2, packed pair (k,k+1).
// Per lane this is two contiguous 8-half runs -> compiler emits ds_load_b128.
__device__ inline v16h load_frag16(const half_t* __restrict__ base, int ld) {
    const int lane = threadIdx.x & 31;
    const int m    = lane & 15;
    const int kb   = (lane >> 4) << 3;        // 0 or 8
    const half_t* row = base + m * ld;
    v16h f;
#pragma unroll
    for (int j = 0; j < 8; ++j) {
        int k = ((j >> 2) << 4) + kb + ((j & 3) << 1);
        f[2 * j]     = row[k];
        f[2 * j + 1] = row[k + 1];
    }
    return f;
}

__device__ inline v8f wmma_f16(v16h a, v16h b, v8f c) {
    return __builtin_amdgcn_wmma_f32_16x16x32_f16(false, a, false, b,
                                                  (short)0, c, false, false);
}

__device__ inline float red_max16(float v) {
    v = fmaxf(v, __shfl_xor(v, 1, 32));
    v = fmaxf(v, __shfl_xor(v, 2, 32));
    v = fmaxf(v, __shfl_xor(v, 4, 32));
    v = fmaxf(v, __shfl_xor(v, 8, 32));
    return v;
}
__device__ inline float red_sum16(float v) {
    v += __shfl_xor(v, 1, 32);
    v += __shfl_xor(v, 2, 32);
    v += __shfl_xor(v, 4, 32);
    v += __shfl_xor(v, 8, 32);
    return v;
}

__device__ inline float gelu_tanh(float h) {
    return h * 0.5f * (1.0f + tanhf(0.79788456f * h * (1.0f + 0.044715f * h * h)));
}

// ----------------------------- elementwise --------------------------------
__global__ __launch_bounds__(256) void k_f32_to_f16(const float* __restrict__ in,
                                                    half_t* __restrict__ out, int n) {
    int i = blockIdx.x * 256 + threadIdx.x;
    if (i < n) out[i] = (half_t)in[i];
}

// LayerNorm over D=1024, one block per row, f16 output
__global__ __launch_bounds__(256) void k_layernorm(const float* __restrict__ x,
                                                   const float* __restrict__ g,
                                                   const float* __restrict__ b,
                                                   half_t* __restrict__ out) {
    __shared__ float red[256];
    const int row = blockIdx.x;
    const int tid = threadIdx.x;
    const float* xr = x + (size_t)row * ND;
    float v0 = xr[tid], v1 = xr[tid + 256], v2 = xr[tid + 512], v3 = xr[tid + 768];
    float s = v0 + v1 + v2 + v3;
    red[tid] = s; __syncthreads();
    for (int o = 128; o > 0; o >>= 1) { if (tid < o) red[tid] += red[tid + o]; __syncthreads(); }
    const float mean = red[0] * (1.0f / ND); __syncthreads();
    float d0 = v0 - mean, d1 = v1 - mean, d2 = v2 - mean, d3 = v3 - mean;
    red[tid] = d0 * d0 + d1 * d1 + d2 * d2 + d3 * d3; __syncthreads();
    for (int o = 128; o > 0; o >>= 1) { if (tid < o) red[tid] += red[tid + o]; __syncthreads(); }
    const float rstd = rsqrtf(red[0] * (1.0f / ND) + 1e-5f);
    half_t* orow = out + (size_t)row * ND;
    orow[tid]       = (half_t)(d0 * rstd * g[tid]       + b[tid]);
    orow[tid + 256] = (half_t)(d1 * rstd * g[tid + 256] + b[tid + 256]);
    orow[tid + 512] = (half_t)(d2 * rstd * g[tid + 512] + b[tid + 512]);
    orow[tid + 768] = (half_t)(d3 * rstd * g[tid + 768] + b[tid + 768]);
}

// RoPE: in f32 [B,L,D] (D = h*64+d), out f16 [B,H,L,HD]
__global__ __launch_bounds__(256) void k_rope(const float* __restrict__ in,
                                              half_t* __restrict__ out) {
    int idx = blockIdx.x * 256 + threadIdx.x;            // 0 .. B*L*D-1
    int c   = idx & (ND - 1);
    int l   = (idx >> 10) & (NL - 1);
    int bb  = idx >> 21;
    int h   = c >> 6;
    int d   = c & 63;
    int dm  = d & 31;
    float inv  = __powf(10000.0f, -(float)(2 * dm) * (1.0f / NHD));
    float ang  = (float)l * inv;
    float cs, sn; __sincosf(ang, &sn, &cs);
    float x1   = in[idx];
    float prt  = (d < 32) ? in[idx + 32] : in[idx - 32];
    float rot  = (d < 32) ? -prt : prt;
    out[(((size_t)bb * NH + h) * NL + l) * NHD + d] = (half_t)(x1 * cs + rot * sn);
}

// ------------------------------- WMMA GEMM --------------------------------
// C[M,N] = act(A_f16[M,K] @ W_f16[K,N] + bias) (+ residual) -> f32 and/or f16
// 256 threads = 8 waves, 128x128 tile, BK=32; wave -> 32x64 (2x4 of 16x16)
#define GLDA 40
__global__ __launch_bounds__(256) void k_gemm(const half_t* __restrict__ A,
                                              const half_t* __restrict__ W,
                                              int M, int N, int K,
                                              const float* __restrict__ bias,
                                              const float* __restrict__ residual,
                                              float* __restrict__ outF,
                                              half_t* __restrict__ outH,
                                              int act) {
    __shared__ half_t As[128][GLDA];
    __shared__ half_t Bt[128][GLDA];   // transposed: Bt[n][k]

    const int tid  = threadIdx.x;
    const int wid  = tid >> 5;
    const int lane = tid & 31;
    const int wm   = wid >> 1;          // 0..3  -> 32-row slice
    const int wn   = wid & 1;           // 0..1  -> 64-col slice
    const int m0   = blockIdx.y * 128;
    const int n0   = blockIdx.x * 128;

    v8f acc[2][4];
#pragma unroll
    for (int i = 0; i < 2; ++i)
#pragma unroll
        for (int j = 0; j < 4; ++j) acc[i][j] = (v8f)0.0f;

    const int arow = tid >> 1, acs = (tid & 1) * 16;
    const int bk   = tid >> 3, bns = (tid & 7) * 16;

    for (int k0 = 0; k0 < K; k0 += 32) {
        const half_t* gA = &A[(size_t)(m0 + arow) * K + k0 + acs];
#if USE_ASYNC_LDS
        // 32B per thread: two async b128 transfers straight into LDS
        async_b128(gA,     &As[arow][acs]);
        async_b128(gA + 8, &As[arow][acs + 8]);
#else
#pragma unroll
        for (int i = 0; i < 16; ++i) As[arow][acs + i] = gA[i];
#endif
#pragma unroll
        for (int i = 0; i < 16; ++i)
            Bt[bns + i][bk] = W[(size_t)(k0 + bk) * N + n0 + bns + i];
        if (k0 + 32 < K) {
            __builtin_prefetch(&A[(size_t)(m0 + arow) * K + k0 + 32 + acs], 0, 1);
            __builtin_prefetch(&W[(size_t)(k0 + 32 + bk) * N + n0 + bns], 0, 1);
        }
#if USE_ASYNC_LDS
        async_wait0();
#endif
        __syncthreads();

        v16h aF[2], bF[4];
#pragma unroll
        for (int mt = 0; mt < 2; ++mt)
            aF[mt] = load_frag16(&As[wm * 32 + mt * 16][0], GLDA);
#pragma unroll
        for (int nt = 0; nt < 4; ++nt)
            bF[nt] = load_frag16(&Bt[wn * 64 + nt * 16][0], GLDA);
#pragma unroll
        for (int mt = 0; mt < 2; ++mt)
#pragma unroll
            for (int nt = 0; nt < 4; ++nt)
                acc[mt][nt] = wmma_f16(aF[mt], bF[nt], acc[mt][nt]);
        __syncthreads();
    }

    // epilogue (C/D layout: VGPR r -> rows r / r+8 across two 16-lane halves)
    const int c    = lane & 15;
    const int half_ = lane >> 4;
#pragma unroll
    for (int mt = 0; mt < 2; ++mt)
#pragma unroll
        for (int nt = 0; nt < 4; ++nt)
#pragma unroll
            for (int r = 0; r < 8; ++r) {
                int row = m0 + wm * 32 + mt * 16 + r + 8 * half_;
                int col = n0 + wn * 64 + nt * 16 + c;
                float v = acc[mt][nt][r] + bias[col];
                if (act) v = gelu_tanh(v);
                if (residual) v += residual[(size_t)row * N + col];
                size_t o = (size_t)row * N + col;
                if (outF) outF[o] = v;
                if (outH) outH[o] = (half_t)v;
            }
}

// --------------------------- flash attention ------------------------------
// grid: (L/128, B*H); 8 waves x 16 query rows; key blocks of 32 in LDS.
#define KLD 72
#define VLD 40
#define PLD 40
__global__ __launch_bounds__(256) void k_attention(const half_t* __restrict__ qh, // [B,H,L,HD]
                                                   const half_t* __restrict__ kh, // [B,H,L,HD]
                                                   const half_t* __restrict__ vh, // [B,L,D]
                                                   half_t* __restrict__ oh) {     // [B,L,D]
    __shared__ half_t Kl[32][KLD];       // [key][d]
    __shared__ half_t Vt[64][VLD];       // [d][key]
    __shared__ half_t Pl[8][16][PLD];    // per-wave P staging

    const int tid   = threadIdx.x;
    const int wid   = tid >> 5;
    const int lane  = tid & 31;
    const int half_ = lane >> 4;
    const int c     = lane & 15;
    const int bh    = blockIdx.y;
    const int b     = bh >> 4;
    const int h     = bh & 15;
    const int qbase = blockIdx.x * 128;
    const int qw    = qbase + wid * 16;          // this wave's first query row
    const int qend  = qbase + 127;

    const half_t* qbaseP = qh + (((size_t)b * NH + h) * NL + qw) * NHD;
    v16h aQ0 = load_frag16(qbaseP, NHD);
    v16h aQ1 = load_frag16(qbaseP + 32, NHD);

    v8f   oacc[4];
    float mrow[8], lrow[8];
#pragma unroll
    for (int t = 0; t < 4; ++t) oacc[t] = (v8f)0.0f;
#pragma unroll
    for (int r = 0; r < 8; ++r) { mrow[r] = -3.0e38f; lrow[r] = 0.0f; }

    const int krow = tid >> 3, kcs = (tid & 7) * 8;

    for (int kbase = 0; kbase <= qend; kbase += 32) {
        // stage K [32x64] and V^T [64x32]
        const half_t* gK =
            &kh[(((size_t)b * NH + h) * NL + kbase + krow) * NHD + kcs];
#if USE_ASYNC_LDS
        async_b128(gK, &Kl[krow][kcs]);     // 16B per thread, ASYNCcnt path
#else
#pragma unroll
        for (int i = 0; i < 8; ++i) Kl[krow][kcs + i] = gK[i];
#endif
#pragma unroll
        for (int i = 0; i < 8; ++i) {
            int idx = tid * 8 + i;          // 0..2047
            int key = idx >> 6, d = idx & 63;
            Vt[d][key] = vh[((size_t)b * NL + kbase + key) * ND + h * NHD + d];
        }
#if USE_ASYNC_LDS
        async_wait0();
#endif
        __syncthreads();

        if (kbase <= qw + 15) {             // not fully masked for this wave
            // S = Q K^T * scale, two 16-key subtiles
            v8f s[2];
#pragma unroll
            for (int t = 0; t < 2; ++t) {
                v16h bK0 = load_frag16(&Kl[t * 16][0],  KLD);
                v16h bK1 = load_frag16(&Kl[t * 16][32], KLD);
                v8f z = (v8f)0.0f;
                z = wmma_f16(aQ0, bK0, z);
                z = wmma_f16(aQ1, bK1, z);
                s[t] = z;
            }
            // scale + causal mask
#pragma unroll
            for (int t = 0; t < 2; ++t)
#pragma unroll
                for (int r = 0; r < 8; ++r) {
                    int row = qw + r + 8 * half_;
                    int col = kbase + t * 16 + c;
                    float v = s[t][r] * 0.125f;
                    s[t][r] = (col > row) ? -3.0e38f : v;
                }
            // online softmax
            float scl[8];
#pragma unroll
            for (int r = 0; r < 8; ++r) {
                float mx = red_max16(fmaxf(s[0][r], s[1][r]));
                float nm = fmaxf(mrow[r], mx);
                float sc = __expf(mrow[r] - nm);
                float p0 = __expf(s[0][r] - nm);
                float p1 = __expf(s[1][r] - nm);
                s[0][r] = p0; s[1][r] = p1;
                lrow[r] = lrow[r] * sc + red_sum16(p0 + p1);
                mrow[r] = nm;
                scl[r]  = sc;
            }
#pragma unroll
            for (int t = 0; t < 4; ++t)
#pragma unroll
                for (int r = 0; r < 8; ++r) oacc[t][r] *= scl[r];
            // stage P (C-layout -> LDS -> A-layout)
#pragma unroll
            for (int r = 0; r < 8; ++r) {
                Pl[wid][r + 8 * half_][c]      = (half_t)s[0][r];
                Pl[wid][r + 8 * half_][16 + c] = (half_t)s[1][r];
            }
            v16h aP = load_frag16(&Pl[wid][0][0], PLD);
#pragma unroll
            for (int t = 0; t < 4; ++t) {
                v16h bV = load_frag16(&Vt[t * 16][0], VLD);
                oacc[t] = wmma_f16(aP, bV, oacc[t]);
            }
        }
        __syncthreads();
    }

    float invl[8];
#pragma unroll
    for (int r = 0; r < 8; ++r) invl[r] = 1.0f / lrow[r];
#pragma unroll
    for (int t = 0; t < 4; ++t)
#pragma unroll
        for (int r = 0; r < 8; ++r) {
            int row = qw + r + 8 * half_;
            oh[((size_t)b * NL + row) * ND + h * NHD + t * 16 + c] =
                (half_t)(oacc[t][r] * invl[r]);
        }
}

// ------------------------------- launcher ---------------------------------
extern "C" void kernel_launch(void* const* d_in, const int* in_sizes, int n_in,
                              void* d_out, int out_size, void* d_ws, size_t ws_size,
                              hipStream_t stream) {
    const float* x     = (const float*)d_in[0];
    const float* Wq    = (const float*)d_in[2];
    const float* bq    = (const float*)d_in[3];
    const float* Wk    = (const float*)d_in[4];
    const float* bk    = (const float*)d_in[5];
    const float* Wv    = (const float*)d_in[6];
    const float* bv    = (const float*)d_in[7];
    const float* Wo    = (const float*)d_in[8];
    const float* bo    = (const float*)d_in[9];
    const float* ln1g  = (const float*)d_in[10];
    const float* ln1b  = (const float*)d_in[11];
    const float* W1    = (const float*)d_in[12];
    const float* b1    = (const float*)d_in[13];
    const float* W2    = (const float*)d_in[14];
    const float* b2    = (const float*)d_in[15];
    const float* ln2g  = (const float*)d_in[16];
    const float* ln2b  = (const float*)d_in[17];
    float* out = (float*)d_out;

    char* ws = (char*)d_ws;
    size_t off = 0;
    auto alloc = [&](size_t bytes) { char* p = ws + off; off = (off + bytes + 255) & ~(size_t)255; return p; };

    half_t* y1h  = (half_t*)alloc((size_t)NROWS * ND * 2);     // LN1 out (reused for LN2)
    half_t* Wqh  = (half_t*)alloc((size_t)ND * ND * 2);
    half_t* Wkh  = (half_t*)alloc((size_t)ND * ND * 2);
    half_t* Wvh  = (half_t*)alloc((size_t)ND * ND * 2);
    half_t* Woh  = (half_t*)alloc((size_t)ND * ND * 2);
    half_t* W1h  = (half_t*)alloc((size_t)ND * NDF * 2);
    half_t* W2h  = (half_t*)alloc((size_t)NDF * ND * 2);
    float*  ftmp = (float*)alloc((size_t)NROWS * ND * 4);      // q/k pre-rope f32
    half_t* qhb  = (half_t*)alloc((size_t)NROWS * ND * 2);     // [B,H,L,HD]
    half_t* khb  = (half_t*)alloc((size_t)NROWS * ND * 2);
    half_t* vhb  = (half_t*)alloc((size_t)NROWS * ND * 2);     // [B,L,D]
    half_t* ohb  = (half_t*)alloc((size_t)NROWS * ND * 2);     // attention out
    float*  x1   = (float*)alloc((size_t)NROWS * ND * 4);      // residual 1
    half_t* hbuf = (half_t*)alloc((size_t)NROWS * NDF * 2);    // MLP hidden

    const int DD = ND * ND;
    k_f32_to_f16<<<(DD + 255) / 256, 256, 0, stream>>>(Wq, Wqh, DD);
    k_f32_to_f16<<<(DD + 255) / 256, 256, 0, stream>>>(Wk, Wkh, DD);
    k_f32_to_f16<<<(DD + 255) / 256, 256, 0, stream>>>(Wv, Wvh, DD);
    k_f32_to_f16<<<(DD + 255) / 256, 256, 0, stream>>>(Wo, Woh, DD);
    k_f32_to_f16<<<(ND * NDF + 255) / 256, 256, 0, stream>>>(W1, W1h, ND * NDF);
    k_f32_to_f16<<<(NDF * ND + 255) / 256, 256, 0, stream>>>(W2, W2h, NDF * ND);

    // LN1
    k_layernorm<<<NROWS, 256, 0, stream>>>(x, ln1g, ln1b, y1h);

    dim3 gDD(ND / 128, NROWS / 128);     // (8, 32)
    // Q
    k_gemm<<<gDD, 256, 0, stream>>>(y1h, Wqh, NROWS, ND, ND, bq, nullptr, ftmp, nullptr, 0);
    k_rope<<<(NROWS * ND) / 256, 256, 0, stream>>>(ftmp, qhb);
    // K
    k_gemm<<<gDD, 256, 0, stream>>>(y1h, Wkh, NROWS, ND, ND, bk, nullptr, ftmp, nullptr, 0);
    k_rope<<<(NROWS * ND) / 256, 256, 0, stream>>>(ftmp, khb);
    // V (straight to f16, [B,L,D])
    k_gemm<<<gDD, 256, 0, stream>>>(y1h, Wvh, NROWS, ND, ND, bv, nullptr, nullptr, vhb, 0);

    // attention
    k_attention<<<dim3(NL / 128, NB * NH), 256, 0, stream>>>(qhb, khb, vhb, ohb);

    // output projection + residual -> x1
    k_gemm<<<gDD, 256, 0, stream>>>(ohb, Woh, NROWS, ND, ND, bo, x, x1, nullptr, 0);

    // LN2 (reuse y1h)
    k_layernorm<<<NROWS, 256, 0, stream>>>(x1, ln2g, ln2b, y1h);

    // MLP
    dim3 gDF(NDF / 128, NROWS / 128);    // (32, 32)
    k_gemm<<<gDF, 256, 0, stream>>>(y1h, W1h, NROWS, NDF, ND, b1, nullptr, nullptr, hbuf, 1);
    k_gemm<<<gDD, 256, 0, stream>>>(hbuf, W2h, NROWS, ND, NDF, b2, x1, out, nullptr, 0);
}
